// LSTMModel_19679540150706
// MI455X (gfx1250) — compile-verified, matrix operand order checked by reading, not run on hardware
//
#include <hip/hip_runtime.h>

// LSTM: B=2048, T=512, input=4, H=128, out=3.
// One workgroup (8 waves) per 16-row batch tile. Wave w owns hidden columns
// [16w, 16w+16) and computes the i/f/g/o gate quadruple for them: cell update
// entirely in registers. The 20 weight WMMA fragments (bf16) are hoisted into
// VGPRs for the whole time loop; LDS only carries the double-buffered A
// operand ([h | x | 0], bf16), so each step is: 10 ds_load (A) + 20 WMMA +
// 8 ds_store (h) + 1 barrier.

typedef __attribute__((ext_vector_type(16))) __bf16 v16bf;
typedef __attribute__((ext_vector_type(8)))  float  v8f;

#define T_STEPS 512
#define HID     128
#define GATES   512           // 4*H
#define KDIM    160           // 128 (h) + 4 (x) + 28 zero pad -> 5 chunks of K=32
#define NCHUNK  5
#define BTILE   16
#define ABUF    (BTILE * KDIM)   // one A buffer, in halves

__device__ __forceinline__ unsigned short f2bf(float f) {
  unsigned int u = __float_as_uint(f);
  return (unsigned short)((u + 0x7FFFu + ((u >> 16) & 1u)) >> 16); // RNE
}
__device__ __forceinline__ float bf2f(unsigned short h) {
  return __uint_as_float(((unsigned int)h) << 16);
}

#if __has_builtin(__builtin_amdgcn_tanhf)
// gfx1250 has v_tanh_f32: tanh in 1 trans op; sigmoid = 0.5*tanh(x/2)+0.5
__device__ __forceinline__ float ftanh_(float x) { return __builtin_amdgcn_tanhf(x); }
__device__ __forceinline__ float fsig(float x)   { return fmaf(0.5f, __builtin_amdgcn_tanhf(0.5f * x), 0.5f); }
#else
__device__ __forceinline__ float fsig(float x)   { return 1.f / (1.f + __expf(-x)); }
__device__ __forceinline__ float ftanh_(float x) { float e = __expf(2.f * x); return 1.f - 2.f / (e + 1.f); }
#endif

union FragU { uint4 q[2]; v16bf v; };

// 16-bit operand fragment, 16x32 tile, row-major [row][K] bf16 with stride KDIM.
// Lane striping per ISA: VGPR0-3 = K {0..7 | 8..15 for lanes>=16}, VGPR4-7 = +16.
__device__ __forceinline__ v16bf load_frag(const unsigned short* base, int row,
                                           int kbase, int lane) {
  const unsigned short* p = base + row * KDIM + kbase + ((lane & 16) ? 8 : 0);
  FragU u;
  u.q[0] = *(const uint4*)(p);        // K +0..7   (packed pairs)
  u.q[1] = *(const uint4*)(p + 16);   // K +16..23 (packed pairs)
  return u.v;
}

__global__ __launch_bounds__(256) void
lstm_wmma_kernel(const float* __restrict__ x,
                 const float* __restrict__ W_ih,
                 const float* __restrict__ W_hh,
                 const float* __restrict__ b_ih,
                 const float* __restrict__ b_hh,
                 const float* __restrict__ W_fc,
                 const float* __restrict__ b_fc,
                 float* __restrict__ out) {
  extern __shared__ __align__(32) unsigned char smem[];
  unsigned short* sW = (unsigned short*)(smem);             // 512*160 bf16 = 160 KB
  unsigned short* sA = (unsigned short*)(smem + 163840);    // 2 * 16*160 bf16 ping-pong

  const int tid  = threadIdx.x;
  const int lane = tid & 31;
  const int wave = tid >> 5;
  const int b0   = blockIdx.x * BTILE;
  const int ncol = (wave << 4) + (lane & 15);   // hidden column this lane owns

  // ---- setup: fused weight matrix W_cat[g][k] (bf16) in LDS ----
  for (int idx = tid; idx < GATES * KDIM; idx += 256) {
    int g = idx / KDIM, k = idx - g * KDIM;
    float v = 0.f;
    if (k < HID)           v = W_hh[g * HID + k];
    else if (k < HID + 4)  v = W_ih[g * 4 + (k - HID)];
    sW[idx] = f2bf(v);
  }
  // init both A buffers (zero pad cols stay zero forever); x[t=0] into buf0
  for (int i = tid; i < 2 * ABUF; i += 256) {
    int r = (i < ABUF) ? i : (i - ABUF);
    int m = r / KDIM, k = r - m * KDIM;
    unsigned short v = 0;
    if (i < ABUF && k >= HID && k < HID + 4)
      v = f2bf(x[((size_t)(b0 + m) * T_STEPS + 0) * 4 + (k - HID)]);
    sA[i] = v;
  }
  // per-lane gate biases (gate order i,f,g,o; W row g = gt*128 + ncol)
  const float bi  = b_ih[0 * HID + ncol] + b_hh[0 * HID + ncol];
  const float bf_ = b_ih[1 * HID + ncol] + b_hh[1 * HID + ncol];
  const float bg  = b_ih[2 * HID + ncol] + b_hh[2 * HID + ncol];
  const float bo  = b_ih[3 * HID + ncol] + b_hh[3 * HID + ncol];
  v8f creg = { 0.f, 0.f, 0.f, 0.f, 0.f, 0.f, 0.f, 0.f };   // cell state in regs
  __syncthreads();

  // ---- hoist all 20 weight fragments into registers for the whole loop ----
  v16bf breg[4][NCHUNK];
#pragma unroll
  for (int gt = 0; gt < 4; ++gt)
#pragma unroll
    for (int c = 0; c < NCHUNK; ++c)
      breg[gt][c] = load_frag(sW, gt * HID + ncol, c * 32, lane);

  const int mbase = (lane & 16) ? 8 : 0;
  const int arow  = lane & 15;

  // ---- time loop: one barrier per step ----
  for (int t = 0; t < T_STEPS; ++t) {
    const unsigned short* Ar = sA + (t & 1) * ABUF;
    unsigned short*       Aw = sA + ((t + 1) & 1) * ABUF;

    v8f acc0 = { bi, bi, bi, bi, bi, bi, bi, bi };
    v8f acc1 = { bf_, bf_, bf_, bf_, bf_, bf_, bf_, bf_ };
    v8f acc2 = { bg, bg, bg, bg, bg, bg, bg, bg };
    v8f acc3 = { bo, bo, bo, bo, bo, bo, bo, bo };
#pragma unroll
    for (int c = 0; c < NCHUNK; ++c) {
      v16bf a = load_frag(Ar, arow, c * 32, lane);
      acc0 = __builtin_amdgcn_wmma_f32_16x16x32_bf16(false, a, false, breg[0][c],
                                                     (short)0, acc0, false, false);
      acc1 = __builtin_amdgcn_wmma_f32_16x16x32_bf16(false, a, false, breg[1][c],
                                                     (short)0, acc1, false, false);
      acc2 = __builtin_amdgcn_wmma_f32_16x16x32_bf16(false, a, false, breg[2][c],
                                                     (short)0, acc2, false, false);
      acc3 = __builtin_amdgcn_wmma_f32_16x16x32_bf16(false, a, false, breg[3][c],
                                                     (short)0, acc3, false, false);
    }

    // cell update fully in registers; D layout: reg r -> M = r + mbase, N = ncol
#pragma unroll
    for (int r = 0; r < 8; ++r) {
      float gi = fsig(acc0[r]);
      float gf = fsig(acc1[r]);
      float gg = ftanh_(acc2[r]);
      float go = fsig(acc3[r]);
      float cv = gf * creg[r] + gi * gg;
      creg[r] = cv;
      float h  = go * ftanh_(cv);
      Aw[(mbase + r) * KDIM + ncol] = f2bf(h);   // h feeds next step's A
    }
    // stage x[t+1] into the write buffer; prefetch x[t+2]
    if (t + 1 < T_STEPS && tid < BTILE * 4) {
      int m = tid >> 2, d = tid & 3;
      const float* xp = x + ((size_t)(b0 + m) * T_STEPS + (t + 1)) * 4 + d;
      Aw[m * KDIM + HID + d] = f2bf(*xp);
      if (t + 2 < T_STEPS) __builtin_prefetch(xp + 4, 0, 1);
    }
    __syncthreads();
  }

  // ---- final FC: h_T lives in buffer (T&1)==0 as bf16 ----
  if (tid < BTILE * 3) {
    int m = tid / 3, o = tid - m * 3;
    float s = b_fc[o];
    const unsigned short* hr = sA + m * KDIM;   // buffer 0
    const float* wr = W_fc + o * HID;
#pragma unroll 8
    for (int j = 0; j < HID; ++j) s += bf2f(hr[j]) * wr[j];
    out[(size_t)(b0 + m) * 3 + o] = s;
  }
}

extern "C" void kernel_launch(void* const* d_in, const int* in_sizes, int n_in,
                              void* d_out, int out_size, void* d_ws, size_t ws_size,
                              hipStream_t stream) {
  const float* x    = (const float*)d_in[0];
  const float* W_ih = (const float*)d_in[1];
  const float* W_hh = (const float*)d_in[2];
  const float* b_ih = (const float*)d_in[3];
  const float* b_hh = (const float*)d_in[4];
  const float* W_fc = (const float*)d_in[5];
  const float* b_fc = (const float*)d_in[6];
  float* out = (float*)d_out;

  const size_t smem = 163840 + 2 * ABUF * sizeof(unsigned short); // 160K W + 10K A
  lstm_wmma_kernel<<<dim3(2048 / BTILE), dim3(256), smem, stream>>>(
      x, W_ih, W_hh, b_ih, b_hh, W_fc, b_fc, out);
}
